// LinearMemoryAttention_64115271794736
// MI455X (gfx1250) — matrix operand order repaired and checked by reading, not compiled, not run on hardware
//
#include <hip/hip_runtime.h>
#include <hip/hip_bf16.h>
#include <cstdint>

// ---------------------------------------------------------------------------
// Linear attention on MI455X (gfx1250):
//   - all GEMMs via v_wmma_f32_16x16x32_f16 (f32 accumulate)
//   - global->LDS staging via global_load_async_to_lds_b128 (ASYNCcnt),
//     2-deep double buffering, branch-free steady-state loop
//   - kv reduction split-K with f32 atomics; norm fused into apply kernel
// ---------------------------------------------------------------------------

typedef _Float16 v8h  __attribute__((ext_vector_type(8)));
typedef _Float16 v16h __attribute__((ext_vector_type(16)));
typedef float    v4f  __attribute__((ext_vector_type(4)));
typedef float    v8f  __attribute__((ext_vector_type(8)));
typedef int      v4i  __attribute__((ext_vector_type(4)));

#define NH    16      // attention heads
#define NKVH  4       // kv heads
#define HD    128     // head dim
#define HID   2048    // hidden size
#define SEQ   4096
#define NB    2
#define LDT   40      // LDS tile row stride in halves (32 + 8 pad, 16B aligned)
#define KVSPL 32      // split-K factor for kv reduction

#if defined(__has_builtin)
#if __has_builtin(__builtin_amdgcn_global_load_async_to_lds_b128)
#define ASYNC_LDS 1
#endif
#endif
#ifndef ASYNC_LDS
#define ASYNC_LDS 0
#endif

// ----- staging helpers ------------------------------------------------------

__device__ __forceinline__ void stage16B(_Float16* lds, const _Float16* g) {
#if ASYNC_LDS
  __builtin_amdgcn_global_load_async_to_lds_b128(
      (__attribute__((address_space(1))) v4i*)(_Float16*)g,
      (__attribute__((address_space(3))) v4i*)lds, 0, 0);
#else
  *(v8h*)lds = *(const v8h*)g;
#endif
}

template <int N>
__device__ __forceinline__ void async_wait() {
#if ASYNC_LDS
#if __has_builtin(__builtin_amdgcn_s_wait_asynccnt)
  __builtin_amdgcn_s_wait_asynccnt(N);
#else
  asm volatile("s_wait_asynccnt %0" ::"i"(N) : "memory");
#endif
#endif
}

// issue one 128x32 A tile + 128x32 B tile (per-thread: 2x16B each side)
__device__ __forceinline__ void issue_tile(_Float16* Ab, _Float16* Bb,
                                           const _Float16* gA, const _Float16* gB,
                                           int off, int k) {
  stage16B(Ab + off,     gA + k);
  stage16B(Ab + off + 8, gA + k + 8);
  stage16B(Bb + off,     gB + k);
  stage16B(Bb + off + 8, gB + k + 8);
}

// Load 16 consecutive f32 from global, convert to f16, store 32B into LDS.
__device__ __forceinline__ void cvt_store16(_Float16* dst, const float* g) {
  v4f f0 = ((const v4f*)g)[0], f1 = ((const v4f*)g)[1],
      f2 = ((const v4f*)g)[2], f3 = ((const v4f*)g)[3];
  v8h h0, h1;
#pragma unroll
  for (int i = 0; i < 4; ++i) {
    h0[i] = (_Float16)f0[i]; h0[i + 4] = (_Float16)f1[i];
    h1[i] = (_Float16)f2[i]; h1[i + 4] = (_Float16)f3[i];
  }
  ((v8h*)dst)[0] = h0;
  ((v8h*)dst)[1] = h1;
}

// ----- WMMA fragment helpers (CDNA5 16x16x32 f16 layouts, wave32) -----------

__device__ __forceinline__ v16h cat8(v8h lo, v8h hi) {
  v16h r;
#pragma unroll
  for (int i = 0; i < 8; ++i) { r[i] = lo[i]; r[i + 8] = hi[i]; }
  return r;
}

__device__ __forceinline__ v16h frag_a(const _Float16* As, int rowBase, int lane) {
  const int l16 = lane & 15, hf = lane >> 4;
  const _Float16* p = As + (size_t)(rowBase + l16) * LDT + hf * 8;
  return cat8(*(const v8h*)p, *(const v8h*)(p + 16));
}

__device__ __forceinline__ v16h frag_b(const _Float16* Bs, int colBase, int lane) {
  const int l16 = lane & 15, hf = lane >> 4;
  const _Float16* p = Bs + (size_t)(colBase + l16) * LDT + hf * 8;
  return cat8(*(const v8h*)p, *(const v8h*)(p + 16));
}

__device__ __forceinline__ void zero_acc(v8f (&acc)[4][2]) {
#pragma unroll
  for (int i = 0; i < 4; ++i)
#pragma unroll
    for (int j = 0; j < 2; ++j)
#pragma unroll
      for (int k = 0; k < 8; ++k) acc[i][j][k] = 0.0f;
}

// Wave computes a 64(M) x 32(N) tile: 4x2 accumulators, one BK=32 step.
__device__ __forceinline__ void wmma_tile(v8f (&acc)[4][2], const _Float16* As,
                                          const _Float16* Bs, int wm, int wn, int lane) {
  v16h a[4], b[2];
#pragma unroll
  for (int i = 0; i < 4; ++i) a[i] = frag_a(As, wm * 64 + i * 16, lane);
#pragma unroll
  for (int j = 0; j < 2; ++j) b[j] = frag_b(Bs, wn * 32 + j * 16, lane);
#pragma unroll
  for (int i = 0; i < 4; ++i)
#pragma unroll
    for (int j = 0; j < 2; ++j)
      acc[i][j] = __builtin_amdgcn_wmma_f32_16x16x32_f16(
          false, a[i], false, b[j], (short)0, acc[i][j], false, false);
}

__device__ __forceinline__ float elu_p1(float x) {
  return x > 0.0f ? x + 1.0f : __expf(x);
}

// ---------------------------------------------------------------------------
// Kernel 0: f32 -> f16 conversion (exact size, n % 8 == 0)
// ---------------------------------------------------------------------------
__global__ __launch_bounds__(256) void cvt_kernel(const float* __restrict__ src,
                                                  _Float16* __restrict__ dst, int n8) {
  const int i = blockIdx.x * 256 + threadIdx.x;
  if (i >= n8) return;
  v4f a = ((const v4f*)src)[i * 2], b = ((const v4f*)src)[i * 2 + 1];
  v8h h;
#pragma unroll
  for (int j = 0; j < 4; ++j) { h[j] = (_Float16)a[j]; h[j + 4] = (_Float16)b[j]; }
  ((v8h*)dst)[i] = h;
}

// ---------------------------------------------------------------------------
// Kernel 1: fused QKV projection (f16 in, async double-buffered staging).
// grid = (64 Mtiles, 24 Ntiles), block 256 (8 waves).
// ---------------------------------------------------------------------------
__global__ __launch_bounds__(256) void qkv_proj_kernel(
    const _Float16* __restrict__ hidden, const _Float16* __restrict__ Wqkv,
    const float* __restrict__ bq, const float* __restrict__ bk,
    const float* __restrict__ bv,
    _Float16* __restrict__ sq, _Float16* __restrict__ sk,
    _Float16* __restrict__ vv) {
  __shared__ alignas(16) _Float16 As[2][128 * LDT];
  __shared__ alignas(16) _Float16 Bs[2][128 * LDT];

  const int tid = threadIdx.x, lane = tid & 31, w = tid >> 5;
  const int wm = w >> 2, wn = w & 3;
  const int m0 = blockIdx.x * 128;
  const int n0 = blockIdx.y * 128;

  const float* bias; _Float16* out; int ldo, c0; bool act;
  if (n0 < HID)            { bias = bq + n0;              out = sq; ldo = NH * HD;   c0 = n0;              act = true;  }
  else if (n0 < HID + 512) { bias = bk + (n0 - HID);      out = sk; ldo = NKVH * HD; c0 = n0 - HID;        act = true;  }
  else                     { bias = bv + (n0 - HID - 512); out = vv; ldo = NKVH * HD; c0 = n0 - HID - 512; act = false; }

  v8f acc[4][2];
  zero_acc(acc);

  const int row = tid >> 1, cg = tid & 1;
  const int off = row * LDT + cg * 16;
  const _Float16* gA = hidden + (size_t)(m0 + row) * HID + cg * 16;
  const _Float16* gB = Wqkv + (size_t)(n0 + row) * HID + cg * 16;

  // software pipeline: prologue issue, branch-free steady state, peeled tail
  issue_tile(As[0], Bs[0], gA, gB, off, 0);
  int cur = 0;
  for (int k0 = 0; k0 < HID - 32; k0 += 32) {
    issue_tile(As[cur ^ 1], Bs[cur ^ 1], gA, gB, off, k0 + 32);
    async_wait<4>();
    __syncthreads();
    wmma_tile(acc, As[cur], Bs[cur], wm, wn, lane);
    __syncthreads();
    cur ^= 1;
  }
  async_wait<0>();
  __syncthreads();
  wmma_tile(acc, As[cur], Bs[cur], wm, wn, lane);

  const int l16 = lane & 15, hf = lane >> 4;
  const float bb0 = bias[wn * 32 + l16];
  const float bb1 = bias[wn * 32 + 16 + l16];
  if (act) {
#pragma unroll
    for (int ms = 0; ms < 4; ++ms) {
      _Float16* p = out + (size_t)(m0 + wm * 64 + ms * 16 + hf * 8) * ldo +
                    c0 + wn * 32 + l16;
#pragma unroll
      for (int r = 0; r < 8; ++r) {
        p[0]  = (_Float16)elu_p1(acc[ms][0][r] + bb0);
        p[16] = (_Float16)elu_p1(acc[ms][1][r] + bb1);
        p += ldo;
      }
    }
  } else {
#pragma unroll
    for (int ms = 0; ms < 4; ++ms) {
      _Float16* p = out + (size_t)(m0 + wm * 64 + ms * 16 + hf * 8) * ldo +
                    c0 + wn * 32 + l16;
#pragma unroll
      for (int r = 0; r < 8; ++r) {
        p[0]  = (_Float16)(acc[ms][0][r] + bb0);
        p[16] = (_Float16)(acc[ms][1][r] + bb1);
        p += ldo;
      }
    }
  }
}

// ---------------------------------------------------------------------------
// Kernel 2: split-K  kvT[e,d] += sum_s sk[s,d]*v[s,e],  k_sum[d] += sum_s sk
// grid = (NB*NKVH, KVSPL); f32 atomics into kvf/ksum (pre-zeroed).
// ---------------------------------------------------------------------------
__global__ __launch_bounds__(256) void kv_kernel(
    const _Float16* __restrict__ sk, const _Float16* __restrict__ vv,
    float* __restrict__ kvf, float* __restrict__ ksum) {
  __shared__ alignas(16) _Float16 Ks[128 * LDT];   // [d][s-chunk]
  __shared__ alignas(16) _Float16 Vs[128 * LDT];   // [e][s-chunk]

  const int tid = threadIdx.x, lane = tid & 31, w = tid >> 5;
  const int wm = w >> 2, wn = w & 3;
  const int bk_id = blockIdx.x;            // b*NKVH + kvh
  const int b = bk_id >> 2, kvh = bk_id & 3;
  const int sbeg = blockIdx.y * (SEQ / KVSPL);

  const _Float16* skb = sk + (size_t)b * SEQ * (NKVH * HD) + kvh * HD;
  const _Float16* vb  = vv + (size_t)b * SEQ * (NKVH * HD) + kvh * HD;

  v8f acc[4][2];
  zero_acc(acc);
  float ksacc = 0.0f;

  const int s_i = tid >> 3;                // 0..31
  const int dg  = (tid & 7) * 16;          // 0,16,...,112

  for (int s0 = sbeg; s0 < sbeg + SEQ / KVSPL; s0 += 32) {
    const _Float16* gk = skb + (size_t)(s0 + s_i) * (NKVH * HD) + dg;
    const _Float16* gv = vb  + (size_t)(s0 + s_i) * (NKVH * HD) + dg;
    v8h k0v = ((const v8h*)gk)[0], k1v = ((const v8h*)gk)[1];
    v8h v0v = ((const v8h*)gv)[0], v1v = ((const v8h*)gv)[1];
#pragma unroll
    for (int j = 0; j < 8; ++j) {
      Ks[(dg + j) * LDT + s_i]     = k0v[j];
      Ks[(dg + 8 + j) * LDT + s_i] = k1v[j];
      Vs[(dg + j) * LDT + s_i]     = v0v[j];
      Vs[(dg + 8 + j) * LDT + s_i] = v1v[j];
    }
    __syncthreads();
    if (tid < 128) {
#pragma unroll
      for (int s = 0; s < 32; ++s) ksacc += (float)Ks[tid * LDT + s];
    }
    wmma_tile(acc, Ks, Vs, wm, wn, lane);
    __syncthreads();
  }

  if (tid < 128) atomicAdd(&ksum[bk_id * 128 + tid], ksacc);

  const int l16 = lane & 15, hf = lane >> 4;
#pragma unroll
  for (int ms = 0; ms < 4; ++ms) {
    const int d0 = wm * 64 + ms * 16 + hf * 8;
#pragma unroll
    for (int ns = 0; ns < 2; ++ns) {
      const int e = wn * 32 + ns * 16 + l16;
      float* p = kvf + (size_t)bk_id * HD * HD + (size_t)e * HD + d0;
#pragma unroll
      for (int r = 0; r < 8; ++r) atomicAdd(p + r, acc[ms][ns][r]);
    }
  }
}

// ---------------------------------------------------------------------------
// Kernel 3: attn = (sq @ kvT^T) / (norm + eps), norm computed in-kernel from
// the staged sq tiles (LDS ds_add_f32 reduction against k_sum).
// grid = (32 s-tiles, 32 b*h), block 256.
// ---------------------------------------------------------------------------
__global__ __launch_bounds__(256) void apply_kernel(
    const _Float16* __restrict__ sq, const float* __restrict__ kvf,
    const float* __restrict__ ksum, _Float16* __restrict__ attn) {
  __shared__ alignas(16) _Float16 As[128 * LDT];
  __shared__ alignas(16) _Float16 Bs[128 * LDT];
  __shared__ float normS[128];

  const int tid = threadIdx.x, lane = tid & 31, w = tid >> 5;
  const int wm = w >> 2, wn = w & 3;
  const int bh = blockIdx.y;
  const int b = bh >> 4, h = bh & 15;
  const int s0 = blockIdx.x * 128;

  const _Float16* qb = sq + ((size_t)b * SEQ + s0) * (NH * HD) + h * HD;
  const float* kvb = kvf + (size_t)(b * NKVH + (h >> 2)) * HD * HD;
  const float* ks = ksum + (b * NKVH + (h >> 2)) * HD;

  v8f acc[4][2];
  zero_acc(acc);

  const int row = tid >> 1, cg = tid & 1;
  const int off = row * LDT + cg * 16;
  if (tid < 128) normS[tid] = 0.0f;   // visible after first barrier below

  for (int k0 = 0; k0 < HD; k0 += 32) {
    stage16B(&As[off],     qb + (size_t)row * (NH * HD) + k0 + cg * 16);
    stage16B(&As[off + 8], qb + (size_t)row * (NH * HD) + k0 + cg * 16 + 8);
    cvt_store16(&Bs[off], kvb + (size_t)row * HD + k0 + cg * 16);
    async_wait<0>();
    __syncthreads();
    {  // per-row partial of norm = sq_row . k_sum over this K chunk
      v8h a0 = *(const v8h*)&As[off];
      v8h a1 = *(const v8h*)&As[off + 8];
      const float* kk = ks + k0 + cg * 16;
      float part = 0.0f;
#pragma unroll
      for (int j = 0; j < 8; ++j)
        part += (float)a0[j] * kk[j] + (float)a1[j] * kk[8 + j];
      atomicAdd(&normS[row], part);
    }
    wmma_tile(acc, As, Bs, wm, wn, lane);
    __syncthreads();   // also fences the ds atomics of this chunk
  }

  const int l16 = lane & 15, hf = lane >> 4;
#pragma unroll
  for (int ms = 0; ms < 4; ++ms) {
    const int m = wm * 64 + ms * 16 + hf * 8;
    _Float16* p = attn + ((size_t)b * SEQ + s0 + m) * (NH * HD) + h * HD +
                  wn * 32 + l16;
#pragma unroll
    for (int r = 0; r < 8; ++r) {
      const float inv = 1.0f / (normS[m + r] + 1e-6f);
      p[0]  = (_Float16)(acc[ms][0][r] * inv);
      p[16] = (_Float16)(acc[ms][1][r] * inv);
      p += NH * HD;
    }
  }
}

// ---------------------------------------------------------------------------
// Kernel 4: out = attn @ Wo^T (f32 out), async double-buffered staging.
// grid = (64, 16)
// ---------------------------------------------------------------------------
__global__ __launch_bounds__(256) void out_proj_kernel(
    const _Float16* __restrict__ attn, const _Float16* __restrict__ Wo,
    float* __restrict__ out) {
  __shared__ alignas(16) _Float16 As[2][128 * LDT];
  __shared__ alignas(16) _Float16 Bs[2][128 * LDT];

  const int tid = threadIdx.x, lane = tid & 31, w = tid >> 5;
  const int wm = w >> 2, wn = w & 3;
  const int m0 = blockIdx.x * 128;
  const int n0 = blockIdx.y * 128;

  v8f acc[4][2];
  zero_acc(acc);

  const int row = tid >> 1, cg = tid & 1;
  const int off = row * LDT + cg * 16;
  const _Float16* gA = attn + (size_t)(m0 + row) * HID + cg * 16;
  const _Float16* gB = Wo + (size_t)(n0 + row) * HID + cg * 16;

  issue_tile(As[0], Bs[0], gA, gB, off, 0);
  int cur = 0;
  for (int k0 = 0; k0 < HID - 32; k0 += 32) {
    issue_tile(As[cur ^ 1], Bs[cur ^ 1], gA, gB, off, k0 + 32);
    async_wait<4>();
    __syncthreads();
    wmma_tile(acc, As[cur], Bs[cur], wm, wn, lane);
    __syncthreads();
    cur ^= 1;
  }
  async_wait<0>();
  __syncthreads();
  wmma_tile(acc, As[cur], Bs[cur], wm, wn, lane);

  const int l16 = lane & 15, hf = lane >> 4;
#pragma unroll
  for (int ms = 0; ms < 4; ++ms) {
    float* p = out + (size_t)(m0 + wm * 64 + ms * 16 + hf * 8) * HID + n0 +
               wn * 32 + l16;
#pragma unroll
    for (int r = 0; r < 8; ++r) {
      p[0]  = acc[ms][0][r];
      p[16] = acc[ms][1][r];
      p += HID;
    }
  }
}

// ---------------------------------------------------------------------------
extern "C" void kernel_launch(void* const* d_in, const int* in_sizes, int n_in,
                              void* d_out, int out_size, void* d_ws, size_t ws_size,
                              hipStream_t stream) {
  (void)in_sizes; (void)n_in; (void)out_size; (void)ws_size;

  const float* hidden = (const float*)d_in[0];
  const float* Wq = (const float*)d_in[1];
  const float* bq = (const float*)d_in[2];
  const float* Wk = (const float*)d_in[3];
  const float* bk = (const float*)d_in[4];
  const float* Wv = (const float*)d_in[5];
  const float* bv = (const float*)d_in[6];
  const float* Wo = (const float*)d_in[7];

  char* ws = (char*)d_ws;
  _Float16* hidden_h = (_Float16*)ws; ws += (size_t)NB * SEQ * HID * 2;       // 33.5 MB
  _Float16* Wqkv_h   = (_Float16*)ws; ws += (size_t)(HID + 1024) * HID * 2;   // 12.6 MB
  _Float16* Wo_h     = (_Float16*)ws; ws += (size_t)HID * HID * 2;            //  8.4 MB
  _Float16* sq       = (_Float16*)ws; ws += (size_t)NB * SEQ * NH * HD * 2;   // 33.5 MB
  _Float16* sk       = (_Float16*)ws; ws += (size_t)NB * SEQ * NKVH * HD * 2; //  8.4 MB
  _Float16* vv       = (_Float16*)ws; ws += (size_t)NB * SEQ * NKVH * HD * 2; //  8.4 MB
  _Float16* attn     = (_Float16*)ws; ws += (size_t)NB * SEQ * NH * HD * 2;   // 33.5 MB
  float*    kvf      = (float*)ws;    ws += (size_t)NB * NKVH * HD * HD * 4;  //  512 KB
  float*    ksum     = (float*)ws;    ws += (size_t)NB * NKVH * HD * 4;       //    4 KB

  // f32 -> f16 conversions (hidden + packed [Wq;Wk;Wv] + Wo)
  {
    int n8;
    n8 = NB * SEQ * HID / 8;
    cvt_kernel<<<dim3((n8 + 255) / 256), 256, 0, stream>>>(hidden, hidden_h, n8);
    n8 = HID * HID / 8;
    cvt_kernel<<<dim3((n8 + 255) / 256), 256, 0, stream>>>(Wq, Wqkv_h, n8);
    n8 = 512 * HID / 8;
    cvt_kernel<<<dim3((n8 + 255) / 256), 256, 0, stream>>>(
        Wk, Wqkv_h + (size_t)HID * HID, n8);
    cvt_kernel<<<dim3((n8 + 255) / 256), 256, 0, stream>>>(
        Wv, Wqkv_h + (size_t)(HID + 512) * HID, n8);
    n8 = HID * HID / 8;
    cvt_kernel<<<dim3((n8 + 255) / 256), 256, 0, stream>>>(Wo, Wo_h, n8);
  }

  // zero split-K accumulators (graph-capture legal)
  (void)hipMemsetAsync(kvf, 0,
                       (size_t)NB * NKVH * HD * HD * 4 + (size_t)NB * NKVH * HD * 4,
                       stream);

  qkv_proj_kernel<<<dim3(64, 24), 256, 0, stream>>>(hidden_h, Wqkv_h, bq, bk, bv,
                                                    sq, sk, vv);
  kv_kernel<<<dim3(8, KVSPL), 256, 0, stream>>>(sk, vv, kvf, ksum);
  apply_kernel<<<dim3(32, 32), 256, 0, stream>>>(sq, kvf, ksum, attn);
  out_proj_kernel<<<dim3(64, 16), 256, 0, stream>>>(attn, Wo_h, (float*)d_out);
}
